// InstructionAwareScorer_29222957482361
// MI455X (gfx1250) — matrix operand order repaired and checked by reading, not compiled, not run on hardware
//
#include <hip/hip_runtime.h>
#include <hip/hip_bf16.h>
#include <math.h>

// ---------------------------------------------------------------------------
// Problem constants (fixed by the reference): B=8, L=1024, N=4096, D=1024
// ---------------------------------------------------------------------------
constexpr int B_ = 8;
constexpr int L_ = 1024;
constexpr int N_ = 4096;
constexpr int D_ = 1024;
constexpr float SCALE = 0.03125f;   // 1/sqrt(1024), TEMPERATURE=1

typedef __attribute__((ext_vector_type(16))) __bf16          v16bf;
typedef __attribute__((ext_vector_type(8)))  float           v8f;
typedef __attribute__((ext_vector_type(16))) unsigned short  v16u;
typedef __attribute__((ext_vector_type(8)))  unsigned short  v8u;
typedef __attribute__((ext_vector_type(4)))  int             v4i;
// address-space-qualified pointee types for the async-LDS builtin
typedef __attribute__((address_space(1))) v4i as1_v4i;
typedef __attribute__((address_space(3))) v4i as3_v4i;

// ---------------- feature probes -------------------------------------------
#if defined(__has_builtin)
#if __has_builtin(__builtin_amdgcn_perm)
#define HAVE_PERM 1
#endif
#if __has_builtin(__builtin_amdgcn_global_load_async_to_lds_b128)
#define HAVE_ASYNC 1
#endif
#endif

// pack two floats into one dword of bf16 (lo | hi<<16), round-half-up
__device__ __forceinline__ unsigned int pack2bf(float lo, float hi) {
    unsigned int a = __builtin_bit_cast(unsigned int, lo) + 0x8000u;
    unsigned int b = __builtin_bit_cast(unsigned int, hi) + 0x8000u;
#ifdef HAVE_PERM
    return __builtin_amdgcn_perm(b, a, 0x07060302);  // {b[3],b[2],a[3],a[2]}
#else
    return (a >> 16) | (b & 0xFFFF0000u);
#endif
}

// raw (unscaled) f32x8 -> bf16x8
__device__ __forceinline__ uint4 cvt8r(float4 a, float4 b) {
    uint4 r;
    r.x = pack2bf(a.x, a.y);
    r.y = pack2bf(a.z, a.w);
    r.z = pack2bf(b.x, b.y);
    r.w = pack2bf(b.z, b.w);
    return r;
}

__device__ __forceinline__ void wait_async0() {
#if defined(__has_builtin)
#if __has_builtin(__builtin_amdgcn_s_wait_asynccnt)
    __builtin_amdgcn_s_wait_asynccnt(0);
    return;
#endif
#endif
#if defined(__AMDGCN__) && defined(HAVE_ASYNC)
    asm volatile("s_wait_asynccnt 0" ::: "memory");
#endif
}

// ---------------------------------------------------------------------------
// K1: per-row inverse L2 norm (one block per row of length D)
// ---------------------------------------------------------------------------
__global__ __launch_bounds__(256)
void rownorm_kernel(const float* __restrict__ x, float* __restrict__ inv, int d) {
    const int row = blockIdx.x;
    const float* p = x + (size_t)row * d;
    float s = 0.f;
    for (int i = threadIdx.x; i < d; i += 256) { float v = p[i]; s += v * v; }
    #pragma unroll
    for (int o = 16; o > 0; o >>= 1) s += __shfl_down(s, o, 32);
    __shared__ float red[8];
    const int lane = threadIdx.x & 31, w = threadIdx.x >> 5;
    if (lane == 0) red[w] = s;
    __syncthreads();
    if (threadIdx.x == 0) {
        float t = 0.f;
        #pragma unroll
        for (int i = 0; i < 8; ++i) t += red[i];
        inv[row] = 1.0f / fmaxf(sqrtf(t), 1e-12f);
    }
}

// ---------------------------------------------------------------------------
// K2: b = beta*mask / (sum_l beta*mask + 1e-8), one block per batch
// ---------------------------------------------------------------------------
__global__ __launch_bounds__(256)
void bnorm_kernel(const float* __restrict__ beta, const int* __restrict__ mask,
                  float* __restrict__ bn) {
    const int b = blockIdx.x;
    float s = 0.f;
    for (int i = threadIdx.x; i < L_; i += 256) {
        float m = mask[b * L_ + i] ? 1.f : 0.f;
        s += beta[b * L_ + i] * m;
    }
    #pragma unroll
    for (int o = 16; o > 0; o >>= 1) s += __shfl_down(s, o, 32);
    __shared__ float red[8];
    __shared__ float total;
    const int lane = threadIdx.x & 31, w = threadIdx.x >> 5;
    if (lane == 0) red[w] = s;
    __syncthreads();
    if (threadIdx.x == 0) {
        float t = 0.f;
        #pragma unroll
        for (int i = 0; i < 8; ++i) t += red[i];
        total = t + 1e-8f;
    }
    __syncthreads();
    const float inv = 1.0f / total;
    for (int i = threadIdx.x; i < L_; i += 256) {
        float m = mask[b * L_ + i] ? 1.f : 0.f;
        bn[b * L_ + i] = beta[b * L_ + i] * m * inv;
    }
}

// ---------------------------------------------------------------------------
// K0: zero a float buffer
// ---------------------------------------------------------------------------
__global__ void zero_kernel(float* __restrict__ p, int n) {
    int i = blockIdx.x * blockDim.x + threadIdx.x;
    if (i < n) p[i] = 0.f;
}

// ---------------------------------------------------------------------------
// K3: tiled bf16 WMMA GEMM on RAW tokens; the L2 norms (rank-1 scaling) and
//   the 1/sqrt(D) scale are folded into the exp epilogue:
//     logit[l,n] = ia_l * ib_n * (raw_t . raw_v)
//   Block tile 64(L) x 128(N), K-step 32, ping-pong LDS, 1 barrier/iter.
//   8 waves = 2(M) x 4(N); each wave owns a 2x2 grid of 16x16 accumulators
//   -> 4 WMMA per wave per K-step, each A/B fragment reused twice.
//   Writes E = exp(logit/32)*mask (unnormalized softmax numerator) to outA.
// ---------------------------------------------------------------------------
#define BM 64
#define BN 128
#define BK 32

#ifdef HAVE_ASYNC
// ======================== ASYNC-DMA variant ================================
#define PF 36   // f32 pitch: 144B rows, 16B-aligned, conflict-free frag reads

template <int ROWS>
__device__ __forceinline__ void issue_tile_async(const float* gbase, float* lbase,
                                                 int k0, int tid) {
    #pragma unroll
    for (int rep = 0; rep < ROWS / 32; ++rep) {   // ROWS*32 f32 = ROWS*8 slots
        const int s   = tid + rep * 256;
        const int row = s >> 3;
        const int q   = s & 7;
        const float* g = gbase + (size_t)row * D_ + k0 + q * 4;
        float*       l = lbase + row * PF + q * 4;
        __builtin_amdgcn_global_load_async_to_lds_b128(
            (as1_v4i*)g, (as3_v4i*)l, 0, 0);
    }
}

__global__ __launch_bounds__(256)
void gemm_exp_kernel(const float* __restrict__ text, const float* __restrict__ vis,
                     const float* __restrict__ invT, const float* __restrict__ invV,
                     const int*   __restrict__ mask, float* __restrict__ outA) {
    __shared__ float Af[2][BM * PF];
    __shared__ float Bf[2][BN * PF];
    __shared__ float iTs[BM], iVs[BN], msk[BM];

    const int bidx = blockIdx.z;
    const int l0   = blockIdx.y * BM;
    const int n0   = blockIdx.x * BN;

    const float* Ap = text + ((size_t)bidx * L_ + l0) * D_;
    const float* Bp = vis  + ((size_t)bidx * N_ + n0) * D_;

    const int t    = threadIdx.x;
    const int lane = t & 31;
    const int wave = t >> 5;
    const int wm   = wave >> 2;    // 0..1
    const int wn   = wave & 3;     // 0..3
    const int half = lane >> 4;    // 0/1
    const int lm   = lane & 15;

    if (t < BM) {
        iTs[t] = invT[bidx * L_ + l0 + t];
        msk[t] = mask[bidx * L_ + l0 + t] ? 1.f : 0.f;
    } else if (t < BM + BN) {
        iVs[t - BM] = invV[bidx * N_ + n0 + (t - BM)];
    }

    v8f c00 = {}, c01 = {}, c10 = {}, c11 = {};

    issue_tile_async<BM>(Ap, &Af[0][0], 0, t);
    issue_tile_async<BN>(Bp, &Bf[0][0], 0, t);
    wait_async0();
    __syncthreads();

    const int r0  = wm * 32 + lm;     // A rows (two M-subtiles)
    const int r1  = r0 + 16;
    const int nc0 = wn * 32 + lm;     // B cols (two N-subtiles)
    const int nc1 = nc0 + 16;

    #pragma unroll 2
    for (int k0 = 0; k0 < D_; k0 += BK) {
        const int cur = (k0 / BK) & 1;
        const int nxt = cur ^ 1;
        const bool more = (k0 + BK) < D_;

        if (more) {  // fire-and-forget DMA for next tile, overlaps WMMA
            issue_tile_async<BM>(Ap, &Af[nxt][0], k0 + BK, t);
            issue_tile_async<BN>(Bp, &Bf[nxt][0], k0 + BK, t);
        }

        // --- fragments per CDNA5 16-bit layouts (raw f32 -> bf16) ----------
        // A 16x32 MxK: lanes 0-15 row M=lane; elems 0-7: K=8*half+e,
        //              elems 8-15: K=16+8*half+(e-8).
        const float* pr0 = &Af[cur][r0 * PF];
        const float* pr1 = &Af[cur][r1 * PF];
        const float* pb0 = &Bf[cur][nc0 * PF];
        const float* pb1 = &Bf[cur][nc1 * PF];

        union { v16u u; uint4 q[2]; } a0f, a1f, b0f, b1f;
        a0f.q[0] = cvt8r(*(const float4*)&pr0[8 * half],
                         *(const float4*)&pr0[8 * half + 4]);
        a0f.q[1] = cvt8r(*(const float4*)&pr0[16 + 8 * half],
                         *(const float4*)&pr0[16 + 8 * half + 4]);
        a1f.q[0] = cvt8r(*(const float4*)&pr1[8 * half],
                         *(const float4*)&pr1[8 * half + 4]);
        a1f.q[1] = cvt8r(*(const float4*)&pr1[16 + 8 * half],
                         *(const float4*)&pr1[16 + 8 * half + 4]);
        // B 32x16 KxN: lane n=lane%16; elems 0-7: K=8*half+e, elems 8-15: K=16+8*half+(e-8)
        b0f.q[0] = cvt8r(*(const float4*)&pb0[8 * half],
                         *(const float4*)&pb0[8 * half + 4]);
        b0f.q[1] = cvt8r(*(const float4*)&pb0[16 + 8 * half],
                         *(const float4*)&pb0[16 + 8 * half + 4]);
        b1f.q[0] = cvt8r(*(const float4*)&pb1[8 * half],
                         *(const float4*)&pb1[8 * half + 4]);
        b1f.q[1] = cvt8r(*(const float4*)&pb1[16 + 8 * half],
                         *(const float4*)&pb1[16 + 8 * half + 4]);

        const v16bf a0 = __builtin_bit_cast(v16bf, a0f.u);
        const v16bf a1 = __builtin_bit_cast(v16bf, a1f.u);
        const v16bf b0 = __builtin_bit_cast(v16bf, b0f.u);
        const v16bf b1 = __builtin_bit_cast(v16bf, b1f.u);

        c00 = __builtin_amdgcn_wmma_f32_16x16x32_bf16(false, a0, false, b0,
                                                      (short)0, c00, false, false);
        c01 = __builtin_amdgcn_wmma_f32_16x16x32_bf16(false, a0, false, b1,
                                                      (short)0, c01, false, false);
        c10 = __builtin_amdgcn_wmma_f32_16x16x32_bf16(false, a1, false, b0,
                                                      (short)0, c10, false, false);
        c11 = __builtin_amdgcn_wmma_f32_16x16x32_bf16(false, a1, false, b1,
                                                      (short)0, c11, false, false);

        if (more) wait_async0();   // our next-tile DMA landed
        __syncthreads();           // publish all waves' LDS writes
    }

    // --- epilogue: E = exp(dot * SCALE*ia*ib) * mask -----------------------
    // C layout: VGPR i, lanes 0-15 -> (M=i, N=lane); lanes 16-31 -> (M=i+8).
    const float sb0 = SCALE * iVs[nc0];
    const float sb1 = SCALE * iVs[nc1];
    #pragma unroll
    for (int ms = 0; ms < 2; ++ms) {
        const int rloc = wm * 32 + ms * 16 + 8 * half;
        v8f ca = ms ? c10 : c00;    // col group 0
        v8f cb = ms ? c11 : c01;    // col group 1
        #pragma unroll
        for (int i = 0; i < 8; ++i) {
            const float ia = iTs[rloc + i];
            const float m  = msk[rloc + i];
            const size_t rowbase = ((size_t)bidx * L_ + l0 + rloc + i) * (size_t)N_;
            outA[rowbase + n0 + nc0] = __expf(ca[i] * sb0 * ia) * m;
            outA[rowbase + n0 + nc1] = __expf(cb[i] * sb1 * ia) * m;
        }
    }
}

#else
// ======================== synchronous fallback =============================
#define PITCH 40   // ushort pitch: 80B rows -> 16B-aligned frags, conflict-free

__global__ __launch_bounds__(256)
void gemm_exp_kernel(const float* __restrict__ text, const float* __restrict__ vis,
                     const float* __restrict__ invT, const float* __restrict__ invV,
                     const int*   __restrict__ mask, float* __restrict__ outA) {
    __shared__ unsigned short At[2][BM * PITCH];
    __shared__ unsigned short Bt[2][BN * PITCH];
    __shared__ float iTs[BM], iVs[BN], msk[BM];

    const int bidx = blockIdx.z;
    const int l0   = blockIdx.y * BM;
    const int n0   = blockIdx.x * BN;

    const float* Ap = text + ((size_t)bidx * L_ + l0) * D_;
    const float* Bp = vis  + ((size_t)bidx * N_ + n0) * D_;

    const int t    = threadIdx.x;
    const int lane = t & 31;
    const int wave = t >> 5;
    const int wm   = wave >> 2;
    const int wn   = wave & 3;
    const int half = lane >> 4;
    const int lm   = lane & 15;

    if (t < BM) {
        iTs[t] = invT[bidx * L_ + l0 + t];
        msk[t] = mask[bidx * L_ + l0 + t] ? 1.f : 0.f;
    } else if (t < BM + BN) {
        iVs[t - BM] = invV[bidx * N_ + n0 + (t - BM)];
    }

    // staging: A tile 64x32 (8 f32/thread), B tile 128x32 (16 f32/thread)
    const int arow = t >> 2,  aoff = (t & 3) * 8;   // 64 rows
    const int brow = t >> 1,  boff = (t & 1) * 16;  // 128 rows, 16 f32 each
    const float* rowA = Ap + (size_t)arow * D_ + aoff;
    const float* rowB = Bp + (size_t)brow * D_ + boff;

    v8f c00 = {}, c01 = {}, c10 = {}, c11 = {};

    {
        const float4* pa = (const float4*)rowA;
        const float4* pb = (const float4*)rowB;
        *(uint4*)&At[0][arow * PITCH + aoff]     = cvt8r(pa[0], pa[1]);
        *(uint4*)&Bt[0][brow * PITCH + boff]     = cvt8r(pb[0], pb[1]);
        *(uint4*)&Bt[0][brow * PITCH + boff + 8] = cvt8r(pb[2], pb[3]);
    }
    __syncthreads();

    const int r0  = wm * 32 + lm;
    const int r1  = r0 + 16;
    const int nc0 = wn * 32 + lm;
    const int nc1 = nc0 + 16;

    #pragma unroll 2
    for (int k0 = 0; k0 < D_; k0 += BK) {
        const int cur = (k0 / BK) & 1;
        const int nxt = cur ^ 1;
        const bool more = (k0 + BK) < D_;

        float4 a0v, a1v, b0v, b1v, b2v, b3v;
        if (more) {
            const float4* pa = (const float4*)(rowA + k0 + BK);
            const float4* pb = (const float4*)(rowB + k0 + BK);
            a0v = pa[0]; a1v = pa[1];
            b0v = pb[0]; b1v = pb[1]; b2v = pb[2]; b3v = pb[3];
        }

        union { v16u u; v8u h[2]; } a0f, a1f;
        a0f.h[0] = *(const v8u*)&At[cur][r0 * PITCH + 16 * half];
        a0f.h[1] = *(const v8u*)&At[cur][r0 * PITCH + 16 * half + 8];
        a1f.h[0] = *(const v8u*)&At[cur][r1 * PITCH + 16 * half];
        a1f.h[1] = *(const v8u*)&At[cur][r1 * PITCH + 16 * half + 8];
        v16u bu0 = *(const v16u*)&Bt[cur][nc0 * PITCH + 16 * half];
        v16u bu1 = *(const v16u*)&Bt[cur][nc1 * PITCH + 16 * half];

        const v16bf a0 = __builtin_bit_cast(v16bf, a0f.u);
        const v16bf a1 = __builtin_bit_cast(v16bf, a1f.u);
        const v16bf b0 = __builtin_bit_cast(v16bf, bu0);
        const v16bf b1 = __builtin_bit_cast(v16bf, bu1);

        c00 = __builtin_amdgcn_wmma_f32_16x16x32_bf16(false, a0, false, b0,
                                                      (short)0, c00, false, false);
        c01 = __builtin_amdgcn_wmma_f32_16x16x32_bf16(false, a0, false, b1,
                                                      (short)0, c01, false, false);
        c10 = __builtin_amdgcn_wmma_f32_16x16x32_bf16(false, a1, false, b0,
                                                      (short)0, c10, false, false);
        c11 = __builtin_amdgcn_wmma_f32_16x16x32_bf16(false, a1, false, b1,
                                                      (short)0, c11, false, false);

        if (more) {
            *(uint4*)&At[nxt][arow * PITCH + aoff]     = cvt8r(a0v, a1v);
            *(uint4*)&Bt[nxt][brow * PITCH + boff]     = cvt8r(b0v, b1v);
            *(uint4*)&Bt[nxt][brow * PITCH + boff + 8] = cvt8r(b2v, b3v);
        }
        __syncthreads();
    }

    const float sb0 = SCALE * iVs[nc0];
    const float sb1 = SCALE * iVs[nc1];
    #pragma unroll
    for (int ms = 0; ms < 2; ++ms) {
        const int rloc = wm * 32 + ms * 16 + 8 * half;
        v8f ca = ms ? c10 : c00;
        v8f cb = ms ? c11 : c01;
        #pragma unroll
        for (int i = 0; i < 8; ++i) {
            const float ia = iTs[rloc + i];
            const float m  = msk[rloc + i];
            const size_t rowbase = ((size_t)bidx * L_ + l0 + rloc + i) * (size_t)N_;
            outA[rowbase + n0 + nc0] = __expf(ca[i] * sb0 * ia) * m;
            outA[rowbase + n0 + nc1] = __expf(cb[i] * sb1 * ia) * m;
        }
    }
}
#endif  // HAVE_ASYNC

// ---------------------------------------------------------------------------
// K3b: rowsum[b,l] = sum_n E[b,l,n]  (one block per row; no atomics)
// ---------------------------------------------------------------------------
__global__ __launch_bounds__(256)
void rowsum_kernel(const float* __restrict__ E, float* __restrict__ rowsum) {
    const int row = blockIdx.x;                 // 0 .. B*L-1
    const float* p = E + (size_t)row * N_;
    float s = 0.f;
    for (int i = threadIdx.x; i < N_; i += 256) s += p[i];
    #pragma unroll
    for (int o = 16; o > 0; o >>= 1) s += __shfl_down(s, o, 32);
    __shared__ float red[8];
    const int lane = threadIdx.x & 31, w = threadIdx.x >> 5;
    if (lane == 0) red[w] = s;
    __syncthreads();
    if (threadIdx.x == 0) {
        float t = 0.f;
        #pragma unroll
        for (int i = 0; i < 8; ++i) t += red[i];
        rowsum[row] = t;
    }
}

// ---------------------------------------------------------------------------
// K4: A = E / rowsum;  s_ia[b,n] += sum_l bn[b,l] * A[b,l,n]
// ---------------------------------------------------------------------------
__global__ __launch_bounds__(256)
void finalize_kernel(const float* __restrict__ rowsum, const float* __restrict__ bn,
                     float* __restrict__ outA, float* __restrict__ sia) {
    const int bidx = blockIdx.z;
    const int n    = blockIdx.x * 256 + threadIdx.x;
    const int lch  = blockIdx.y * 128;
    float acc = 0.f;
    size_t base = ((size_t)bidx * L_ + lch) * (size_t)N_ + n;
    for (int l = 0; l < 128; ++l) {
        const int   lg  = lch + l;
        const float rs  = rowsum[bidx * L_ + lg];
        const float inv = (rs > 0.f) ? (1.0f / rs) : 0.f;
        float a = outA[base + (size_t)l * N_] * inv;
        outA[base + (size_t)l * N_] = a;
        acc += bn[bidx * L_ + lg] * a;
    }
    atomicAdd(&sia[bidx * N_ + n], acc);
}

// ---------------------------------------------------------------------------
extern "C" void kernel_launch(void* const* d_in, const int* in_sizes, int n_in,
                              void* d_out, int out_size, void* d_ws, size_t ws_size,
                              hipStream_t stream) {
    const float* text = (const float*)d_in[0];   // (B, L, D) f32
    const float* vis  = (const float*)d_in[1];   // (B, N, D) f32
    const float* beta = (const float*)d_in[2];   // (B, L)    f32
    const int*   mask = (const int*)d_in[3];     // (B, L)    i32

    float* out  = (float*)d_out;
    float* sia  = out;                           // (B, N)
    float* outA = out + (size_t)B_ * N_;         // (B, L, N)

    float* invT   = (float*)d_ws;                // B*L
    float* invV   = invT + B_ * L_;              // B*N
    float* rowsum = invV + B_ * N_;              // B*L
    float* bn     = rowsum + B_ * L_;            // B*L

    rownorm_kernel<<<B_ * L_, 256, 0, stream>>>(text, invT, D_);
    rownorm_kernel<<<B_ * N_, 256, 0, stream>>>(vis,  invV, D_);
    bnorm_kernel<<<B_, 256, 0, stream>>>(beta, mask, bn);
    zero_kernel<<<(B_ * N_ + 255) / 256, 256, 0, stream>>>(sia, B_ * N_);

    dim3 g3(N_ / BN, L_ / BM, B_);
    gemm_exp_kernel<<<g3, 256, 0, stream>>>(text, vis, invT, invV, mask, outA);

    rowsum_kernel<<<B_ * L_, 256, 0, stream>>>(outA, rowsum);

    dim3 g4(N_ / 256, L_ / 128, B_);
    finalize_kernel<<<g4, 256, 0, stream>>>(rowsum, bn, outA, sia);
}
